// UnifiedAttention_78194174591356
// MI455X (gfx1250) — matrix operand order repaired and checked by reading, not compiled, not run on hardware
//
#include <hip/hip_runtime.h>
#include <math.h>

// ---------------------------------------------------------------------------
// Problem constants (from reference): B=2, T=128, P=192, H=8, E=64, D=H*E=512
// ---------------------------------------------------------------------------
#define B_ 2
#define T_ 128
#define P_ 192
#define H_ 8
#define E_ 64
#define D_ 512
#define M_ (B_ * T_ * P_)          // 49152 "rows" (b,t,p)
#define QKV_STRIDE ((size_t)B_ * H_ * P_ * T_ * E_)  // elems per q/k/v plane

typedef _Float16 v8h  __attribute__((ext_vector_type(8)));
typedef _Float16 v16h __attribute__((ext_vector_type(16)));
typedef float    v8f  __attribute__((ext_vector_type(8)));
typedef unsigned int v4u __attribute__((ext_vector_type(4)));
typedef int      v4i  __attribute__((ext_vector_type(4)));
typedef int      v8i  __attribute__((ext_vector_type(8)));

#ifndef __has_builtin
#define __has_builtin(x) 0
#endif
#if __has_builtin(__builtin_amdgcn_tensor_load_to_lds) && \
    __has_builtin(__builtin_amdgcn_s_wait_tensorcnt)
#define HAVE_TDM 1
#else
#define HAVE_TDM 0
#endif

// ---------------------------------------------------------------------------
// WMMA fragment loaders, CDNA5 ISA §7.12.2 layouts.
// A (16x32 f16): lane<16 -> M=lane, K=k0+0..7 / k0+16..23; lanes 16-31 ->
//                K=k0+8..15 / k0+24..31.
// B from transposed [N][K] storage: lane<16 -> N=lane, K=k0+0..7 / k0+16..23;
//                lanes 16-31 -> K=k0+8..15 / k0+24..31.
// ---------------------------------------------------------------------------
__device__ __forceinline__ v16h frag_a(const _Float16* A, int lda, int m, int k0, int lane) {
  int kb = k0 + ((lane & 16) ? 8 : 0);
  const v8h lo = *(const v8h*)(A + (size_t)m * lda + kb);
  const v8h hi = *(const v8h*)(A + (size_t)m * lda + kb + 16);
  v16h r;
#pragma unroll
  for (int i = 0; i < 8; ++i) { r[i] = lo[i]; r[i + 8] = hi[i]; }
  return r;
}

__device__ __forceinline__ v16h frag_b(const _Float16* Bt, int ldb, int n, int k0, int lane) {
  int kb = k0 + ((lane & 16) ? 8 : 0);
  const v8h lo = *(const v8h*)(Bt + (size_t)n * ldb + kb);
  const v8h hi = *(const v8h*)(Bt + (size_t)n * ldb + kb + 16);
  v16h r;
#pragma unroll
  for (int i = 0; i < 8; ++i) { r[i] = lo[i]; r[i + 8] = hi[i]; }
  return r;
}

// One 16x16 f32 tile of A[M,K] * Bt[N,K]^T (used by the attention kernels).
__device__ __forceinline__ v8f wmma_tile(const _Float16* A, int lda,
                                         const _Float16* Bt, int ldb,
                                         int m0, int n0, int K, int lane) {
  v8f acc = {};
  const int mrow = m0 + (lane & 15);
  const int ncol = n0 + (lane & 15);
  for (int k0 = 0; k0 < K; k0 += 32) {
    v16h a = frag_a(A, lda, mrow, k0, lane);
    v16h b = frag_b(Bt, ldb, ncol, k0, lane);
    acc = __builtin_amdgcn_wmma_f32_16x16x32_f16(false, a, false, b,
                                                 (short)0, acc, false, false);
  }
  return acc;
}

#if HAVE_TDM
// ---------------------------------------------------------------------------
// TDM 2-D tile load: tile_k x tile_n f16 elements, row stride `row_stride`
// elements, from `gaddr` (tile start) into LDS byte offset `lds_off`.
// D# per CDNA5 ISA.
// ---------------------------------------------------------------------------
__device__ __forceinline__ void tdm_load_2d(unsigned lds_off, const void* gaddr,
                                            unsigned tile_k, unsigned tile_n,
                                            unsigned tensor_k, unsigned tensor_n,
                                            unsigned row_stride) {
  unsigned long long ga = (unsigned long long)(size_t)gaddr;
  v4u g0;
  g0[0] = 1u;                                            // count=1, user desc
  g0[1] = lds_off;                                       // lds_addr (bytes)
  g0[2] = (unsigned)(ga & 0xffffffffu);                  // global_addr[31:0]
  g0[3] = (unsigned)((ga >> 32) & 0x1ffffffu) | (2u << 30);  // addr[56:32]|type=2
  v8i g1;
  g1[0] = (int)(1u << 16);                               // data_size=1 (2 bytes)
  g1[1] = (int)((tensor_k & 0xffffu) << 16);             // tensor_dim0[15:0]
  g1[2] = (int)(((tensor_k >> 16) & 0xffffu) | ((tensor_n & 0xffffu) << 16));
  g1[3] = (int)(((tensor_n >> 16) & 0xffffu) | ((tile_k & 0xffffu) << 16));
  g1[4] = (int)(tile_n & 0xffffu);                       // tile_dim1 (tile_dim2=0)
  g1[5] = (int)row_stride;                               // tensor_dim0_stride lo
  g1[6] = 0;
  g1[7] = 0;
  v4i z4 = {};
#if defined(__clang_major__) && (__clang_major__ >= 23)
  v8i z8 = {};
  __builtin_amdgcn_tensor_load_to_lds(g0, g1, z4, z4, z8, 0);
#else
  __builtin_amdgcn_tensor_load_to_lds(g0, g1, z4, z4, 0);
#endif
}
#endif

// ---------------------------------------------------------------------------
// Block-tiled GEMM mainloop: 256 threads / 8 waves, C tile = 128x128.
// Wave w owns rows [16w,16w+16); 8 accumulators across 128 N columns.
// DOUBLE-BUFFERED TDM staging: 2x (128x32 A + 128x32 B) = 32 KB LDS.
// Wave 0 prefetches the pair for step s+1, then s_wait_tensorcnt 2 (tensor
// ops complete in order per wave -> <=2 outstanding means step-s pair landed),
// overlapping DMA with the 8 WMMAs of step s.  B-fragment loads are software
// pipelined one iteration ahead of their WMMA.
// ---------------------------------------------------------------------------
__device__ __forceinline__ void gemm128_mainloop(const _Float16* __restrict__ A,
                                                 const _Float16* __restrict__ Bt,
                                                 int Ntot, int mbase, int nbase,
                                                 int Ktot, char* smem, v8f acc[8]) {
  const int tid = threadIdx.x, lane = tid & 31, wave = tid >> 5;
  const int steps = Ktot / 32;
#if HAVE_TDM
  if (tid < 32) {   // scalar-branched away for waves 1..7 (TDM ignores EXEC)
    tdm_load_2d(0u,    A  + (size_t)mbase * Ktot, 32u, 128u,
                (unsigned)Ktot, (unsigned)M_,  (unsigned)Ktot);
    tdm_load_2d(8192u, Bt + (size_t)nbase * Ktot, 32u, 128u,
                (unsigned)Ktot, (unsigned)Ntot, (unsigned)Ktot);
  }
#endif
  for (int s = 0; s < steps; ++s) {
    const int cur = s & 1;
    _Float16* As = (_Float16*)(smem + cur * 16384);
    _Float16* Bs = (_Float16*)(smem + cur * 16384 + 8192);
#if HAVE_TDM
    if (tid < 32) {
      if (s + 1 < steps) {
        const int k1 = (s + 1) * 32;
        const unsigned nxt = (unsigned)(1 - cur) * 16384u;
        tdm_load_2d(nxt,         A  + (size_t)mbase * Ktot + k1, 32u, 128u,
                    (unsigned)Ktot, (unsigned)M_,  (unsigned)Ktot);
        tdm_load_2d(nxt + 8192u, Bt + (size_t)nbase * Ktot + k1, 32u, 128u,
                    (unsigned)Ktot, (unsigned)Ntot, (unsigned)Ktot);
        __builtin_amdgcn_s_wait_tensorcnt((short)2);
      } else {
        __builtin_amdgcn_s_wait_tensorcnt((short)0);
      }
    }
#else
    const int k0 = s * 32;
    for (int i = tid; i < 512; i += 256) {
      int r = i >> 2, c = (i & 3) * 8;
      ((v8h*)As)[i] = *(const v8h*)(A  + (size_t)(mbase + r) * Ktot + k0 + c);
      ((v8h*)Bs)[i] = *(const v8h*)(Bt + (size_t)(nbase + r) * Ktot + k0 + c);
    }
#endif
    __syncthreads();   // staged tile visible to all waves
    v16h af = frag_a(As, 32, wave * 16 + (lane & 15), 0, lane);
    v16h bf = frag_b(Bs, 32, (lane & 15), 0, lane);
#pragma unroll
    for (int nt = 0; nt < 8; ++nt) {
      v16h bn;
      if (nt < 7) bn = frag_b(Bs, 32, (nt + 1) * 16 + (lane & 15), 0, lane);
      acc[nt] = __builtin_amdgcn_wmma_f32_16x16x32_f16(false, af, false, bf,
                                                       (short)0, acc[nt], false, false);
      bf = bn;
    }
    __syncthreads();   // all waves done before next prefetch overwrites buffer
  }
}

// ---------------------------------------------------------------------------
// Conversion kernels: fp32 -> f16, weights transposed to [N][K].
// ---------------------------------------------------------------------------
__global__ void k_cvt_x(const float* __restrict__ x, _Float16* __restrict__ xh, int n) {
  int i = blockIdx.x * 256 + threadIdx.x;
  if (i < n) xh[i] = (_Float16)x[i];
}

// w_qkv_frame [3,D,H,E] -> wt[n=(q,h,e)][k=d], N=1536, K=512
__global__ void k_cvt_wf(const float* __restrict__ w, _Float16* __restrict__ wt) {
  int i = blockIdx.x * 256 + threadIdx.x;
  if (i >= 1536 * 512) return;
  int nn = i >> 9, k = i & 511;
  int q = nn >> 9, he = nn & 511;
  int h = he >> 6, e = he & 63;
  wt[i] = (_Float16)w[(((size_t)q * D_ + k) * H_ + h) * E_ + e];
}

// w_qkv_point [D,H,E] -> wt[n=(h,e)][k=d], N=512, K=512
__global__ void k_cvt_wp(const float* __restrict__ w, _Float16* __restrict__ wt) {
  int i = blockIdx.x * 256 + threadIdx.x;
  if (i >= 512 * 512) return;
  int nn = i >> 9, k = i & 511;
  int h = nn >> 6, e = nn & 63;
  wt[i] = (_Float16)w[((size_t)k * H_ + h) * E_ + e];
}

// w_out [H,E,D] -> wt[n=d][k=(h,e)], N=512, K=512
__global__ void k_cvt_wo(const float* __restrict__ w, _Float16* __restrict__ wt) {
  int i = blockIdx.x * 256 + threadIdx.x;
  if (i >= 512 * 512) return;
  int nn = i >> 9, k = i & 511;          // nn = d, k = h*64+e
  wt[i] = (_Float16)w[(size_t)k * D_ + nn];
}

// ---------------------------------------------------------------------------
// K1: QKV-frame projection.  A = xh [M,512], Bt = wf_t [1536][512].
// Epilogue: +bias[h,e], scatter f16 to qkvf[q][b][h][p][t][e].
// grid (12, 384), block 256, 32 KB dynamic LDS.
// ---------------------------------------------------------------------------
__global__ void k_qkv_frame(const _Float16* __restrict__ xh,
                            const _Float16* __restrict__ wt,
                            const float* __restrict__ bias,
                            _Float16* __restrict__ qkvf) {
  extern __shared__ char smem[];
  v8f acc[8];
#pragma unroll
  for (int i = 0; i < 8; ++i) acc[i] = (v8f){};
  const int mbase = blockIdx.y * 128, nbase = blockIdx.x * 128;
  gemm128_mainloop(xh, wt, 1536, mbase, nbase, D_, smem, acc);
  const int lane = threadIdx.x & 31, wave = threadIdx.x >> 5;
  const int m0 = mbase + wave * 16;
#pragma unroll
  for (int nt = 0; nt < 8; ++nt) {
    int n = nbase + nt * 16 + (lane & 15);
    float bv = bias[n & 511];
    int q = n >> 9, h = (n >> 6) & 7, e = n & 63;
#pragma unroll
    for (int r = 0; r < 8; ++r) {
      int m = m0 + r + ((lane & 16) ? 8 : 0);
      int b = m / (T_ * P_);
      int tp = m % (T_ * P_);
      int t = tp / P_, p = tp % P_;
      size_t idx = ((((size_t)q * B_ + b) * H_ + h) * P_ + p) * ((size_t)T_ * E_) +
                   (size_t)t * E_ + e;
      qkvf[idx] = (_Float16)(acc[nt][r] + bv);
    }
  }
}

// ---------------------------------------------------------------------------
// K2: frame attention.  One block per (b,h,p); 256 threads = 8 waves.
// LDS: Qs[128*64] Ks[128*64] Vt[64*128] f16 + S[128*128] f32 + Wh[128*128] f16
//    = 144 KB dynamic LDS.  Q/K/V staged by TDM (raw V parked in the Wh
// region, transposed into Vt from LDS before Wh is needed).
// ---------------------------------------------------------------------------
__global__ void k_frame_attn(const _Float16* __restrict__ qkvf,
                             _Float16* __restrict__ frame_out) {
  extern __shared__ char smem[];
  _Float16* Qs = (_Float16*)smem;                         // 128x64 @0
  _Float16* Ks = Qs + 128 * 64;                           // 128x64 @16384
  _Float16* Vt = Ks + 128 * 64;                           // 64x128 @32768
  float*    S  = (float*)(smem + 49152);                  // 128x128 f32 @49152
  _Float16* Wh = (_Float16*)(smem + 114688);              // 128x128 f16 @114688

  const int tid = threadIdx.x, lane = tid & 31, wave = tid >> 5;
  const int blk = blockIdx.x;
  const int b = blk / (H_ * P_);
  const int h = (blk / P_) % H_;
  const int p = blk % P_;

  const size_t qbase = (((size_t)b * H_ + h) * P_ + p) * ((size_t)T_ * E_);
  const _Float16* Qg = qkvf + qbase;
  const _Float16* Kg = qkvf + qbase + QKV_STRIDE;
  const _Float16* Vg = qkvf + qbase + 2 * QKV_STRIDE;

#if HAVE_TDM
  if (tid < 32) {
    tdm_load_2d(0u,      Qg, 64u, 128u, 64u, 128u, 64u);
    tdm_load_2d(16384u,  Kg, 64u, 128u, 64u, 128u, 64u);
    tdm_load_2d(114688u, Vg, 64u, 128u, 64u, 128u, 64u);  // raw V -> Wh region
    __builtin_amdgcn_s_wait_tensorcnt((short)0);
  }
  __syncthreads();
  const _Float16* Vraw = Wh;
  for (int i = tid; i < 128 * 64; i += 256) {
    int t = i >> 6, e = i & 63;
    Vt[e * 128 + t] = Vraw[i];
  }
#else
  for (int i = tid; i < 1024; i += 256) {
    ((v8h*)Qs)[i] = ((const v8h*)Qg)[i];
    ((v8h*)Ks)[i] = ((const v8h*)Kg)[i];
  }
  for (int i = tid; i < 128 * 64; i += 256) {
    int t = i >> 6, e = i & 63;
    Vt[e * 128 + t] = Vg[i];
  }
#endif
  __syncthreads();

  // S = Q K^T * scale
  const float scale = 0.125f;  // 1/sqrt(64)
  const int m0 = wave * 16;
  for (int ct = 0; ct < 8; ++ct) {
    v8f acc = wmma_tile(Qs, 64, Ks, 64, m0, ct * 16, 64, lane);
    int n = ct * 16 + (lane & 15);
#pragma unroll
    for (int r = 0; r < 8; ++r) {
      int m = m0 + r + ((lane & 16) ? 8 : 0);
      S[m * 128 + n] = acc[r] * scale;
    }
  }
  __syncthreads();

  // Softmax over key-time axis; one thread per row.  (Overwrites Wh -- raw V
  // already transposed into Vt.)
  if (tid < 128) {
    float mx = -1e30f;
    for (int c = 0; c < 128; ++c) mx = fmaxf(mx, S[tid * 128 + c]);
    float sum = 0.f;
    for (int c = 0; c < 128; ++c) {
      float ev = __expf(S[tid * 128 + c] - mx);
      S[tid * 128 + c] = ev;
      sum += ev;
    }
    float inv = 1.f / sum;
    for (int c = 0; c < 128; ++c) Wh[tid * 128 + c] = (_Float16)(S[tid * 128 + c] * inv);
  }
  __syncthreads();

  // O = W V  -> frame_out[b][s][p][h*64+e]  (f16)
  for (int et = 0; et < 4; ++et) {
    v8f acc = wmma_tile(Wh, 128, Vt, 128, m0, et * 16, 128, lane);
    int e = et * 16 + (lane & 15);
#pragma unroll
    for (int r = 0; r < 8; ++r) {
      int s = m0 + r + ((lane & 16) ? 8 : 0);
      size_t idx = (((size_t)b * T_ + s) * P_ + p) * D_ + h * E_ + e;
      frame_out[idx] = (_Float16)acc[r];
    }
  }
}

// ---------------------------------------------------------------------------
// K3: point projection.  A = frame_out [M,512], Bt = wp_t [512][512].
// Epilogue: +bias, scatter to pproj[b][h][t][p][e].  grid (4, 384), 32 KB LDS.
// ---------------------------------------------------------------------------
__global__ void k_point_proj(const _Float16* __restrict__ fo,
                             const _Float16* __restrict__ wt,
                             const float* __restrict__ bias,
                             _Float16* __restrict__ pproj) {
  extern __shared__ char smem[];
  v8f acc[8];
#pragma unroll
  for (int i = 0; i < 8; ++i) acc[i] = (v8f){};
  const int mbase = blockIdx.y * 128, nbase = blockIdx.x * 128;
  gemm128_mainloop(fo, wt, 512, mbase, nbase, D_, smem, acc);
  const int lane = threadIdx.x & 31, wave = threadIdx.x >> 5;
  const int m0 = mbase + wave * 16;
#pragma unroll
  for (int nt = 0; nt < 8; ++nt) {
    int n = nbase + nt * 16 + (lane & 15);
    float bv = bias[n];
    int h = n >> 6, e = n & 63;
#pragma unroll
    for (int r = 0; r < 8; ++r) {
      int m = m0 + r + ((lane & 16) ? 8 : 0);
      int b = m / (T_ * P_);
      int tp = m % (T_ * P_);
      int t = tp / P_, p = tp % P_;
      size_t idx = (((size_t)b * H_ + h) * T_ + t) * ((size_t)P_ * E_) +
                   (size_t)p * E_ + e;
      pproj[idx] = (_Float16)(acc[nt][r] + bv);
    }
  }
}

// ---------------------------------------------------------------------------
// K4: point attention (q=k=v=proj).  One block per (b,h,t); 384 threads =
// 12 waves, wave = one 16-row tile of P=192.
// LDS: Ps[192*64] + Vt[64*192] f16 + S[192*192] f32 + Wh[192*192] f16 = 264 KB.
// Ps staged by TDM; Vt transposed from LDS.
// ---------------------------------------------------------------------------
__global__ void k_point_attn(const _Float16* __restrict__ pproj,
                             _Float16* __restrict__ point_out) {
  extern __shared__ char smem[];
  _Float16* Ps = (_Float16*)smem;                               // 192x64 @0
  _Float16* Vt = (_Float16*)(smem + 24576);                     // 64x192
  float*    S  = (float*)(smem + 49152);                        // 192x192 f32
  _Float16* Wh = (_Float16*)(smem + 196608);                    // 192x192 f16

  const int tid = threadIdx.x, lane = tid & 31, wave = tid >> 5;
  const int blk = blockIdx.x;
  const int b = blk / (H_ * T_);
  const int h = (blk / T_) % H_;
  const int t = blk % T_;

  const _Float16* Pg = pproj + (((size_t)b * H_ + h) * T_ + t) * ((size_t)P_ * E_);

#if HAVE_TDM
  if (tid < 32) {
    tdm_load_2d(0u, Pg, 64u, 192u, 64u, 192u, 64u);
    __builtin_amdgcn_s_wait_tensorcnt((short)0);
  }
  __syncthreads();
  for (int i = tid; i < 192 * 64; i += 384) {
    int p = i >> 6, e = i & 63;
    Vt[e * 192 + p] = Ps[i];
  }
#else
  for (int i = tid; i < (192 * 64) / 8; i += 384) ((v8h*)Ps)[i] = ((const v8h*)Pg)[i];
  for (int i = tid; i < 192 * 64; i += 384) {
    int p = i >> 6, e = i & 63;
    Vt[e * 192 + p] = Pg[i];
  }
#endif
  __syncthreads();

  const float scale = 0.125f;
  const int m0 = wave * 16;
  for (int ct = 0; ct < 12; ++ct) {
    v8f acc = wmma_tile(Ps, 64, Ps, 64, m0, ct * 16, 64, lane);
    int n = ct * 16 + (lane & 15);
#pragma unroll
    for (int r = 0; r < 8; ++r) {
      int m = m0 + r + ((lane & 16) ? 8 : 0);
      S[m * 192 + n] = acc[r] * scale;
    }
  }
  __syncthreads();

  if (tid < 192) {
    float mx = -1e30f;
    for (int c = 0; c < 192; ++c) mx = fmaxf(mx, S[tid * 192 + c]);
    float sum = 0.f;
    for (int c = 0; c < 192; ++c) {
      float ev = __expf(S[tid * 192 + c] - mx);
      S[tid * 192 + c] = ev;
      sum += ev;
    }
    float inv = 1.f / sum;
    for (int c = 0; c < 192; ++c) Wh[tid * 192 + c] = (_Float16)(S[tid * 192 + c] * inv);
  }
  __syncthreads();

  // O = W V -> point_out[b][t][p][h*64+e]  (f16)
  for (int et = 0; et < 4; ++et) {
    v8f acc = wmma_tile(Wh, 192, Vt, 192, m0, et * 16, 192, lane);
    int e = et * 16 + (lane & 15);
#pragma unroll
    for (int r = 0; r < 8; ++r) {
      int p = m0 + r + ((lane & 16) ? 8 : 0);
      size_t idx = (((size_t)b * T_ + t) * P_ + p) * D_ + h * E_ + e;
      point_out[idx] = (_Float16)acc[r];
    }
  }
}

// ---------------------------------------------------------------------------
// K5: output projection + bias + residual.  A = point_out [M,512],
// Bt = wo_t [512][512].  out f32 = acc + b_out[n] + x[m,n].  grid (4, 384).
// ---------------------------------------------------------------------------
__global__ void k_out_proj(const _Float16* __restrict__ po,
                           const _Float16* __restrict__ wt,
                           const float* __restrict__ b_out,
                           const float* __restrict__ x,
                           float* __restrict__ out) {
  extern __shared__ char smem[];
  v8f acc[8];
#pragma unroll
  for (int i = 0; i < 8; ++i) acc[i] = (v8f){};
  const int mbase = blockIdx.y * 128, nbase = blockIdx.x * 128;
  gemm128_mainloop(po, wt, 512, mbase, nbase, D_, smem, acc);
  const int lane = threadIdx.x & 31, wave = threadIdx.x >> 5;
  const int m0 = mbase + wave * 16;
#pragma unroll
  for (int nt = 0; nt < 8; ++nt) {
    int n = nbase + nt * 16 + (lane & 15);
    float bv = b_out[n];
#pragma unroll
    for (int r = 0; r < 8; ++r) {
      int m = m0 + r + ((lane & 16) ? 8 : 0);
      size_t idx = (size_t)m * D_ + n;
      out[idx] = acc[nt][r] + bv + x[idx];
    }
  }
}

// ---------------------------------------------------------------------------
// Host launcher.  Workspace (f16): xh 48MB (reused as point_out), weights
// ~2.5MB, qkvf 144MB, frame_out 48MB, pproj 48MB -> ~290 MiB total.
// ---------------------------------------------------------------------------
extern "C" void kernel_launch(void* const* d_in, const int* in_sizes, int n_in,
                              void* d_out, int out_size, void* d_ws, size_t ws_size,
                              hipStream_t stream) {
  const float* x    = (const float*)d_in[0];
  const float* w_f  = (const float*)d_in[1];
  const float* b_f  = (const float*)d_in[2];
  const float* w_p  = (const float*)d_in[3];
  const float* b_p  = (const float*)d_in[4];
  const float* w_o  = (const float*)d_in[5];
  const float* b_o  = (const float*)d_in[6];
  float* out = (float*)d_out;

  char* ws = (char*)d_ws;
  size_t off = 0;
  _Float16* xh   = (_Float16*)(ws + off); off += (size_t)M_ * D_ * 2;
  _Float16* wf_t = (_Float16*)(ws + off); off += (size_t)1536 * 512 * 2;
  _Float16* wp_t = (_Float16*)(ws + off); off += (size_t)512 * 512 * 2;
  _Float16* wo_t = (_Float16*)(ws + off); off += (size_t)512 * 512 * 2;
  _Float16* qkvf = (_Float16*)(ws + off); off += 3 * QKV_STRIDE * 2;
  _Float16* fo   = (_Float16*)(ws + off); off += (size_t)M_ * D_ * 2;
  _Float16* pp   = (_Float16*)(ws + off); off += (size_t)B_ * H_ * T_ * P_ * E_ * 2;
  _Float16* po   = xh;  // xh consumed by K1; reuse for point attention output

  // Conversions
  k_cvt_x<<<(M_ * D_ + 255) / 256, 256, 0, stream>>>(x, xh, M_ * D_);
  k_cvt_wf<<<(1536 * 512 + 255) / 256, 256, 0, stream>>>(w_f, wf_t);
  k_cvt_wp<<<(512 * 512 + 255) / 256, 256, 0, stream>>>(w_p, wp_t);
  k_cvt_wo<<<(512 * 512 + 255) / 256, 256, 0, stream>>>(w_o, wo_t);

  // Pipeline
  k_qkv_frame<<<dim3(12, M_ / 128), 256, 32768, stream>>>(xh, wf_t, b_f, qkvf);
  k_frame_attn<<<B_ * H_ * P_, 256, 147456, stream>>>(qkvf, fo);
  k_point_proj<<<dim3(4, M_ / 128), 256, 32768, stream>>>(fo, wp_t, b_p, pp);
  k_point_attn<<<B_ * H_ * T_, 384, 270336, stream>>>(pp, po);
  k_out_proj<<<dim3(4, M_ / 128), 256, 32768, stream>>>(po, wo_t, b_o, x, out);
}
